// AudioLSTMInference_75359496175902
// MI455X (gfx1250) — matrix operand order repaired and checked by reading, not compiled, not run on hardware
//
#include <hip/hip_runtime.h>
#include <hip/hip_bf16.h>
#include <stdint.h>

// Problem constants (match reference)
#define BB    128
#define TT    1000
#define DD    40
#define HH    128
#define CC    8
#define G4H   512                 // 4*H
#define MROWS (BB*TT)             // 128000

typedef __bf16 bf16_t;
typedef __attribute__((ext_vector_type(8)))  bf16_t v8bf;
typedef __attribute__((ext_vector_type(16))) bf16_t v16bf;
typedef __attribute__((ext_vector_type(8)))  float  v8f;

__device__ __forceinline__ unsigned short f2bf(float f) {
    unsigned u = __builtin_bit_cast(unsigned, f);
    u += 0x7fffu + ((u >> 16) & 1u);          // round-to-nearest-even
    return (unsigned short)(u >> 16);
}
__device__ __forceinline__ float bf2f(unsigned short h) {
    return __builtin_bit_cast(float, (unsigned)h << 16);
}
__device__ __forceinline__ float sigm(float x) { return 1.0f / (1.0f + __expf(-x)); }

// ---------------------------------------------------------------------------
// LayerNorm over D=40 -> bf16 rows padded to 64. One wave per row.
// ---------------------------------------------------------------------------
__global__ void ln_kernel(const float* __restrict__ x, const float* __restrict__ g,
                          const float* __restrict__ bta, unsigned short* __restrict__ xout) {
    int wid = threadIdx.x >> 5, lane = threadIdx.x & 31;
    int row = blockIdx.x * 8 + wid;
    if (row >= MROWS) return;
    const float* xr = x + (size_t)row * DD;
    int k2 = lane + 32;
    float x0 = xr[lane];                       // lane < 32 < 40 always valid
    float x1 = (k2 < DD) ? xr[k2] : 0.f;
    float s = x0 + x1, q = x0 * x0 + x1 * x1;
    for (int o = 16; o; o >>= 1) { s += __shfl_xor(s, o, 32); q += __shfl_xor(q, o, 32); }
    float mu  = s * (1.0f / DD);
    float var = q * (1.0f / DD) - mu * mu;
    float rs  = rsqrtf(var + 1e-5f);
    unsigned short* orow = xout + (size_t)row * 64;
    orow[lane] = f2bf((x0 - mu) * rs * g[lane] + bta[lane]);
    orow[k2]   = (k2 < DD) ? f2bf((x1 - mu) * rs * g[k2] + bta[k2]) : (unsigned short)0;
}

// ---------------------------------------------------------------------------
// Convert f32 weights [N,K] row-major -> bf16 [N,Kp], zero padding K..Kp.
// ---------------------------------------------------------------------------
__global__ void wcvt_kernel(const float* __restrict__ src, unsigned short* __restrict__ dst,
                            int N, int K, int Kp) {
    int idx = blockIdx.x * blockDim.x + threadIdx.x;
    if (idx >= N * Kp) return;
    int n = idx / Kp, k = idx - n * Kp;
    dst[idx] = (k < K) ? f2bf(src[n * K + k]) : (unsigned short)0;
}

__global__ void bsum_kernel(const float* __restrict__ a, const float* __restrict__ b,
                            float* __restrict__ o, int n) {
    int i = blockIdx.x * blockDim.x + threadIdx.x;
    if (i < n) o[i] = a[i] + b[i];
}

// ---------------------------------------------------------------------------
// WMMA GEMM: out[M,N] = act( A[M,Kp](bf16,rm) * Bw[N,Kp](bf16,rm)^T + bias )
// One 16x16 tile per wave, 8 waves (=128 N columns) per 256-thread block.
// grid = (M/16, N/128). ACT: 0=none, 1=tanh, 2=relu.
// ---------------------------------------------------------------------------
template <int ACT, bool WF, bool WB>
__global__ void wmma_gemm(const unsigned short* __restrict__ A,
                          const unsigned short* __restrict__ Bw,
                          const float* __restrict__ bias,
                          float* __restrict__ outF,
                          unsigned short* __restrict__ outB,
                          int N, int Kp) {
    int wid = threadIdx.x >> 5, lane = threadIdx.x & 31;
    int l15 = lane & 15, lh = lane >> 4;
    int mbase = blockIdx.x * 16;
    int nt = blockIdx.y * 8 + wid;
    int n  = nt * 16 + l15;

    v8f acc;
    float binit = bias ? bias[n] : 0.f;
    for (int i = 0; i < 8; ++i) acc[i] = binit;

    const unsigned short* arow = A  + (size_t)(mbase + l15) * Kp;
    const unsigned short* brow = Bw + (size_t)n * Kp;

    for (int kb = 0; kb < Kp; kb += 32) {
        union { v16bf v; v8bf h[2]; } ua, ub;
        // A 16x32 layout: lane-half selects K base {0,16}; elems 0..15 contiguous
        ua.h[0] = *(const v8bf*)(arow + kb + 16 * lh);
        ua.h[1] = *(const v8bf*)(arow + kb + 16 * lh + 8);
        // B 32x16 layout: lane-half selects K base {0,16}; n = lane%16
        ub.v = *(const v16bf*)(brow + kb + 16 * lh);
        acc = __builtin_amdgcn_wmma_f32_16x16x32_bf16(false, ua.v, false, ub.v,
                                                      (short)0, acc, false, false);
    }
    for (int i = 0; i < 8; ++i) {
        float v = acc[i];
        if (ACT == 1) v = tanhf(v);
        if (ACT == 2) v = fmaxf(v, 0.f);
        size_t row = (size_t)(mbase + i + 8 * lh);
        if (WF) outF[row * N + n] = v;
        if (WB) outB[row * N + n] = f2bf(v);
    }
}

// ---------------------------------------------------------------------------
// Persistent LSTM recurrence. grid = (8 batch-slices of 16, 2 directions),
// block = 256 (8 waves). Whh is preloaded into 128 VGPRs/wave and stays
// register-resident for all 1000 steps (__launch_bounds__(256,1) gives the
// full VGPR budget; only 16 WGs exist so occupancy is irrelevant).
// Per step: 32 gate tiles (8 waves x 4) of h[16,128] @ Whh^T[128,512] via
// WMMA with C initialized from the precomputed input projections; then
// elementwise gate math with the c-state held in registers.
// ---------------------------------------------------------------------------
__global__ __launch_bounds__(256, 1)
void lstm_rec_kernel(const float* __restrict__ gates_f,
                     const float* __restrict__ gates_b,
                     const unsigned short* __restrict__ whh_f,
                     const unsigned short* __restrict__ whh_b,
                     unsigned short* __restrict__ hout /* [M, 2H] */) {
    __shared__ __align__(16) unsigned short h_sh[16 * HH];   // bf16 h state
    __shared__ float g_sh[16 * G4H];                          // gate preacts

    const int dir = blockIdx.y;
    const float* gates = dir ? gates_b : gates_f;
    const unsigned short* whh = dir ? whh_b : whh_f;
    const int bbase = blockIdx.x * 16;
    const int tid = threadIdx.x;
    const int wid = tid >> 5, lane = tid & 31, l15 = lane & 15, lh = lane >> 4;
    const int j  = tid & 127;           // hidden index owned in elementwise phase
    const int m0 = (tid >> 7) * 8;      // first of 8 owned batch rows

    // Preload this wave's Whh tiles: 4 n-tiles x 4 k-steps x 8 VGPRs = 128 VGPRs.
    v16bf breg[4][4];
#pragma unroll
    for (int q = 0; q < 4; ++q) {
        int n = (wid * 4 + q) * 16 + l15;
        const unsigned short* brow = whh + (size_t)n * HH;
#pragma unroll
        for (int kk = 0; kk < 4; ++kk)
            breg[q][kk] = *(const v16bf*)(brow + kk * 32 + 16 * lh);
    }

    float c_reg[8];
#pragma unroll
    for (int r = 0; r < 8; ++r) c_reg[r] = 0.f;
    for (int idx = tid; idx < 16 * HH; idx += 256) h_sh[idx] = 0;
    __syncthreads();

    for (int s = 0; s < TT; ++s) {
        int t = dir ? (TT - 1 - s) : s;
#pragma unroll
        for (int q = 0; q < 4; ++q) {
            int n = (wid * 4 + q) * 16 + l15;
            v8f acc;
#pragma unroll
            for (int i = 0; i < 8; ++i) {
                int m = i + 8 * lh;
                acc[i] = gates[((size_t)(bbase + m) * TT + t) * G4H + n];
            }
#pragma unroll
            for (int kk = 0; kk < 4; ++kk) {
                int kb = kk * 32;
                union { v16bf v; v8bf h[2]; } ua;
                ua.h[0] = *(const v8bf*)&h_sh[l15 * HH + kb + 16 * lh];
                ua.h[1] = *(const v8bf*)&h_sh[l15 * HH + kb + 16 * lh + 8];
                acc = __builtin_amdgcn_wmma_f32_16x16x32_bf16(false, ua.v, false, breg[q][kk],
                                                              (short)0, acc, false, false);
            }
#pragma unroll
            for (int i = 0; i < 8; ++i) g_sh[(i + 8 * lh) * G4H + n] = acc[i];
        }
        __syncthreads();
        // gate order i,f,g,o; c = sig(f)*c + sig(i)*tanh(g); h = sig(o)*tanh(c)
#pragma unroll
        for (int r = 0; r < 8; ++r) {
            int m = m0 + r;
            const float* gm = &g_sh[m * G4H];
            float iv = sigm(gm[j]);
            float fv = sigm(gm[HH + j]);
            float gv = tanhf(gm[2 * HH + j]);
            float ov = sigm(gm[3 * HH + j]);
            float cv = fv * c_reg[r] + iv * gv;
            c_reg[r] = cv;
            unsigned short hb = f2bf(ov * tanhf(cv));
            h_sh[m * HH + j] = hb;
            hout[((size_t)(bbase + m) * TT + t) * (2 * HH) + dir * HH + j] = hb;
        }
        __syncthreads();
    }
}

// ---------------------------------------------------------------------------
// scores[row] = dot(S1[row,0:128], w2). One wave per row.
// ---------------------------------------------------------------------------
__global__ void scores_kernel(const float* __restrict__ S1, const float* __restrict__ w2,
                              float* __restrict__ sc) {
    int wid = threadIdx.x >> 5, lane = threadIdx.x & 31;
    int row = blockIdx.x * 8 + wid;
    if (row >= MROWS) return;
    const float* r = S1 + (size_t)row * HH;
    float s = 0.f;
    for (int q = 0; q < 4; ++q) { int k = q * 32 + lane; s += r[k] * w2[k]; }
    for (int o = 16; o; o >>= 1) s += __shfl_xor(s, o, 32);
    if (lane == 0) sc[row] = s;
}

// Softmax over T, in place. One block per batch element.
__global__ void softmax_kernel(float* __restrict__ sc) {
    __shared__ float red[256];
    int b = blockIdx.x, tid = threadIdx.x;
    float* p = sc + (size_t)b * TT;
    float mx = -3.0e38f;
    for (int t = tid; t < TT; t += 256) mx = fmaxf(mx, p[t]);
    red[tid] = mx; __syncthreads();
    for (int o = 128; o > 0; o >>= 1) { if (tid < o) red[tid] = fmaxf(red[tid], red[tid + o]); __syncthreads(); }
    mx = red[0]; __syncthreads();
    float sm = 0.f;
    for (int t = tid; t < TT; t += 256) sm += __expf(p[t] - mx);
    red[tid] = sm; __syncthreads();
    for (int o = 128; o > 0; o >>= 1) { if (tid < o) red[tid] += red[tid + o]; __syncthreads(); }
    float inv = 1.f / red[0]; __syncthreads();
    for (int t = tid; t < TT; t += 256) p[t] = __expf(p[t] - mx) * inv;
}

// ctx[b,d] = sum_t w[b,t] * h1[b,t,d]. grid=B, block=256 (d = 2H features).
__global__ void ctx_kernel(const float* __restrict__ w, const unsigned short* __restrict__ h1,
                           float* __restrict__ ctxF, unsigned short* __restrict__ ctxB) {
    int b = blockIdx.x, d = threadIdx.x;
    const float* wb = w + (size_t)b * TT;
    const unsigned short* hb = h1 + (size_t)b * TT * (2 * HH) + d;
    float acc = 0.f;
    for (int t = 0; t < TT; ++t) acc += wb[t] * bf2f(hb[(size_t)t * (2 * HH)]);
    ctxF[b * 256 + d] = acc;
    ctxB[b * 256 + d] = f2bf(acc);
}

// Final tiny layer: out[b,c] = b2[c] + hid[b,:] . W2[c,:]. 1024 threads.
__global__ void head2_kernel(const float* __restrict__ hid, const float* __restrict__ W2,
                             const float* __restrict__ b2, float* __restrict__ out) {
    int tid = blockIdx.x * blockDim.x + threadIdx.x;
    if (tid >= BB * CC) return;
    int b = tid >> 3, c = tid & 7;
    float acc = b2[c];
    const float* hr = hid + b * HH;
    const float* wr = W2 + c * HH;
    for (int k = 0; k < HH; ++k) acc += hr[k] * wr[k];
    out[tid] = acc;
}

// ---------------------------------------------------------------------------
extern "C" void kernel_launch(void* const* d_in, const int* in_sizes, int n_in,
                              void* d_out, int out_size, void* d_ws, size_t ws_size,
                              hipStream_t stream) {
    const float* x      = (const float*)d_in[0];
    const float* ln_g   = (const float*)d_in[1];
    const float* ln_b   = (const float*)d_in[2];
    // order in dict: f0, b0, f1, b1
    const float* Wih[4] = {(const float*)d_in[3],  (const float*)d_in[7],
                           (const float*)d_in[11], (const float*)d_in[15]};
    const float* Whh[4] = {(const float*)d_in[4],  (const float*)d_in[8],
                           (const float*)d_in[12], (const float*)d_in[16]};
    const float* bih[4] = {(const float*)d_in[5],  (const float*)d_in[9],
                           (const float*)d_in[13], (const float*)d_in[17]};
    const float* bhh[4] = {(const float*)d_in[6],  (const float*)d_in[10],
                           (const float*)d_in[14], (const float*)d_in[18]};
    const float* attn_W1 = (const float*)d_in[19];
    const float* attn_b1 = (const float*)d_in[20];
    const float* attn_w2 = (const float*)d_in[21];
    const float* fc_W1   = (const float*)d_in[22];
    const float* fc_b1   = (const float*)d_in[23];
    const float* fc_W2   = (const float*)d_in[24];
    const float* fc_b2   = (const float*)d_in[25];
    float* out = (float*)d_out;

    // Workspace carve-up (256B aligned)
    char* wsp = (char*)d_ws; size_t off = 0;
    auto carve = [&](size_t bytes) -> void* {
        void* p = wsp + off; off += (bytes + 255) & ~(size_t)255; return p;
    };
    unsigned short* x0     = (unsigned short*)carve((size_t)MROWS * 64 * 2);
    float*          gatesF = (float*)carve((size_t)MROWS * G4H * 4);
    float*          gatesB = (float*)carve((size_t)MROWS * G4H * 4);
    unsigned short* h0     = (unsigned short*)carve((size_t)MROWS * 256 * 2);
    unsigned short* h1     = (unsigned short*)carve((size_t)MROWS * 256 * 2);
    unsigned short* wih0f  = (unsigned short*)carve(512 * 64 * 2);
    unsigned short* wih0b  = (unsigned short*)carve(512 * 64 * 2);
    unsigned short* wih1f  = (unsigned short*)carve(512 * 256 * 2);
    unsigned short* wih1b  = (unsigned short*)carve(512 * 256 * 2);
    unsigned short* whhb[4];
    for (int i = 0; i < 4; ++i) whhb[i] = (unsigned short*)carve(512 * 128 * 2);
    unsigned short* attnW1b = (unsigned short*)carve(128 * 256 * 2);
    unsigned short* fcW1b   = (unsigned short*)carve(128 * 256 * 2);
    float* biasc[4];
    for (int i = 0; i < 4; ++i) biasc[i] = (float*)carve(512 * 4);
    float*          ctxF = (float*)carve(128 * 256 * 4);
    unsigned short* ctxB = (unsigned short*)carve(128 * 256 * 2);
    float*          hidF = (float*)carve(128 * 128 * 4);
    // Aliases (gate buffers are dead after layer-1 recurrence)
    float* S1 = gatesF;   // [MROWS,128] tanh(attn) activations
    float* sc = gatesB;   // [MROWS] scores -> softmax weights

    // --- weight / bias prep ---
    auto cvt = [&](const float* s, unsigned short* d, int N, int K, int Kp) {
        int tot = N * Kp;
        wcvt_kernel<<<(tot + 255) / 256, 256, 0, stream>>>(s, d, N, K, Kp);
    };
    cvt(Wih[0], wih0f, 512, 40, 64);
    cvt(Wih[1], wih0b, 512, 40, 64);
    cvt(Wih[2], wih1f, 512, 256, 256);
    cvt(Wih[3], wih1b, 512, 256, 256);
    for (int i = 0; i < 4; ++i) cvt(Whh[i], whhb[i], 512, 128, 128);
    cvt(attn_W1, attnW1b, 128, 256, 256);
    cvt(fc_W1,   fcW1b,   128, 256, 256);
    for (int i = 0; i < 4; ++i)
        bsum_kernel<<<2, 256, 0, stream>>>(bih[i], bhh[i], biasc[i], 512);

    // --- LayerNorm -> bf16 padded input ---
    ln_kernel<<<MROWS / 8, 256, 0, stream>>>(x, ln_g, ln_b, x0);

    // --- Layer 0: input projections + recurrence ---
    {
        dim3 g(MROWS / 16, G4H / 128);
        wmma_gemm<0, true, false><<<g, 256, 0, stream>>>(x0, wih0f, biasc[0], gatesF, nullptr, G4H, 64);
        wmma_gemm<0, true, false><<<g, 256, 0, stream>>>(x0, wih0b, biasc[1], gatesB, nullptr, G4H, 64);
        lstm_rec_kernel<<<dim3(BB / 16, 2), 256, 0, stream>>>(gatesF, gatesB, whhb[0], whhb[1], h0);
    }
    // --- Layer 1 ---
    {
        dim3 g(MROWS / 16, G4H / 128);
        wmma_gemm<0, true, false><<<g, 256, 0, stream>>>(h0, wih1f, biasc[2], gatesF, nullptr, G4H, 256);
        wmma_gemm<0, true, false><<<g, 256, 0, stream>>>(h0, wih1b, biasc[3], gatesB, nullptr, G4H, 256);
        lstm_rec_kernel<<<dim3(BB / 16, 2), 256, 0, stream>>>(gatesF, gatesB, whhb[2], whhb[3], h1);
    }
    // --- Attention pooling ---
    wmma_gemm<1, true, false><<<dim3(MROWS / 16, 1), 256, 0, stream>>>(h1, attnW1b, attn_b1, S1, nullptr, HH, 256);
    scores_kernel<<<MROWS / 8, 256, 0, stream>>>(S1, attn_w2, sc);
    softmax_kernel<<<BB, 256, 0, stream>>>(sc);
    ctx_kernel<<<BB, 256, 0, stream>>>(sc, h1, ctxF, ctxB);

    // --- Classifier head ---
    wmma_gemm<2, true, false><<<dim3(BB / 16, 1), 256, 0, stream>>>(ctxB, fcW1b, fc_b1, hidF, nullptr, HH, 256);
    head2_kernel<<<1, 1024, 0, stream>>>(hidF, fc_W2, fc_b2, out);
    (void)in_sizes; (void)n_in; (void)out_size; (void)ws_size;
}
